// SupConLoss_6356551598243
// MI455X (gfx1250) — matrix operand order repaired and checked by reading, not compiled, not run on hardware
//
#include <hip/hip_runtime.h>
#include <hip/hip_bf16.h>

// ---- CDNA5 WMMA / TDM vector types (wave32) ----
typedef __attribute__((ext_vector_type(16))) __bf16 v16bf;
typedef __attribute__((ext_vector_type(8)))  __bf16 v8bf;
typedef __attribute__((ext_vector_type(4)))  __bf16 v4bf;
typedef __attribute__((ext_vector_type(8)))  float  v8f;
typedef __attribute__((ext_vector_type(4)))  unsigned int u32x4;
typedef __attribute__((ext_vector_type(8)))  int          i32x8;
typedef __attribute__((ext_vector_type(4)))  int          i32x4;

#define NROWS 8192
#define DIM   512
#define INV_T (1.0f / 0.07f)
#define TILE_ELEMS (16 * DIM)   // one B tile: 16 rows x 512 bf16 = 16 KB, contiguous

#if __has_builtin(__builtin_amdgcn_tensor_load_to_lds) && \
    __has_builtin(__builtin_amdgcn_s_wait_tensorcnt)
#define USE_TDM 1
#else
#define USE_TDM 0
#endif

// ---------------------------------------------------------------------------
// Kernel 1: L2-normalize each row (fp32 in) and emit bf16 normalized features.
// ---------------------------------------------------------------------------
__global__ void __launch_bounds__(256)
supcon_normalize(const float* __restrict__ feat, __bf16* __restrict__ fbf) {
  int gwave = (blockIdx.x * blockDim.x + threadIdx.x) >> 5;
  int lane  = threadIdx.x & 31;
  if (gwave >= NROWS) return;

  const float* row = feat + (size_t)gwave * DIM;
  float4 v[4];
  float ss = 0.f;
#pragma unroll
  for (int i = 0; i < 4; ++i) {
    v[i] = *(const float4*)(row + lane * 4 + i * 128);
    ss += v[i].x * v[i].x + v[i].y * v[i].y + v[i].z * v[i].z + v[i].w * v[i].w;
  }
#pragma unroll
  for (int m = 16; m >= 1; m >>= 1) ss += __shfl_xor(ss, m, 32);

  float inv = 1.0f / fmaxf(sqrtf(ss), 1e-12f);   // F.normalize eps

  __bf16* orow = fbf + (size_t)gwave * DIM;
#pragma unroll
  for (int i = 0; i < 4; ++i) {
    v4bf o;
    o[0] = (__bf16)(v[i].x * inv);
    o[1] = (__bf16)(v[i].y * inv);
    o[2] = (__bf16)(v[i].z * inv);
    o[3] = (__bf16)(v[i].w * inv);
    *(v4bf*)(orow + lane * 4 + i * 128) = o;
  }
}

#if USE_TDM
// Issue a TDM copy of one contiguous 16 KB B tile (Global -> LDS).
// 1-D descriptor: data_size=2B, tile_dim0=8192 elements, type=2 ("image").
__device__ __forceinline__ void tdm_load_tile(unsigned lds_addr,
                                              const __bf16* gsrc) {
  unsigned long long ga = (unsigned long long)(uintptr_t)gsrc;
  u32x4 g0;
  g0[0] = 1u;                                   // count=1 (valid user D#)
  g0[1] = lds_addr;                             // LDS byte address
  g0[2] = (unsigned)(ga & 0xFFFFFFFFu);         // global_addr[31:0]
  g0[3] = (unsigned)((ga >> 32) & 0x01FFFFFFu)  // global_addr[56:32]
        | (2u << 30);                           // type=2
  i32x8 g1;
  g1[0] = (int)(1u << 16);                              // data_size=1 -> 2 bytes
  g1[1] = (int)((TILE_ELEMS & 0xFFFFu) << 16);          // tensor_dim0[15:0]
  g1[2] = (int)(((unsigned)TILE_ELEMS >> 16) | (1u << 16)); // dim0 hi | tensor_dim1=1
  g1[3] = (int)((TILE_ELEMS & 0xFFFFu) << 16);          // tile_dim0 = 8192
  g1[4] = 0;                                            // tile_dim1/2 unused (1-D)
  g1[5] = (int)TILE_ELEMS;                              // tensor_dim0_stride lo32
  g1[6] = 0;
  g1[7] = 0;
#if __clang_major__ >= 23
  __builtin_amdgcn_tensor_load_to_lds(g0, g1, (i32x4)0, (i32x4)0, (i32x8)0, 0);
#else
  __builtin_amdgcn_tensor_load_to_lds(g0, g1, (i32x4)0, (i32x4)0, 0);
#endif
}
#endif

// ---------------------------------------------------------------------------
// Kernel 2: fused  S = f f^T  (bf16 WMMA, f32 acc)  +  SupCon statistics.
// Workgroup = 8 waves, 128 rows (one 16-row block per wave). B tiles are
// staged into LDS once per workgroup (TDM, double-buffered) and consumed by
// all 8 waves. Inner K-loop uses TWO independent WMMA accumulation chains
// and register double-buffered B fragments to hide LDS and XDL latency.
// ---------------------------------------------------------------------------
__global__ void __launch_bounds__(256)
supcon_main(const __bf16* __restrict__ fbf,
            const int* __restrict__ labels,
            float* __restrict__ acc) {
  __shared__ __bf16 ldsB[2][TILE_ELEMS];   // 2 x 16 KB double buffer

  const int wave = threadIdx.x >> 5;
  const int lane = threadIdx.x & 31;
  const int hf   = lane >> 4;      // which 16-lane half
  const int lr   = lane & 15;

  const int rowBlock = blockIdx.x * 8 + wave;   // 0..511
  const int rowBase  = rowBlock * 16;

  // --- A fragments, register resident (K-contiguous 16 elems at 16*hf, matching B) ---
  const __bf16* arow = fbf + (size_t)(rowBase + lr) * DIM;
  v16bf afrag[16];
#pragma unroll
  for (int kc = 0; kc < 16; ++kc) {
    const int k0 = kc * 32;
    v8bf lo = *(const v8bf*)(arow + k0 + 16 * hf);
    v8bf hi = *(const v8bf*)(arow + k0 + 16 * hf + 8);
    v16bf a;
#pragma unroll
    for (int e = 0; e < 8; ++e) { a[e] = lo[e]; a[e + 8] = hi[e]; }
    afrag[kc] = a;
  }

  int rowLab[8];
#pragma unroll
  for (int v = 0; v < 8; ++v) rowLab[v] = labels[rowBase + v + 8 * hf];

  float sumExp[8], sumS[8], cnt[8];
#pragma unroll
  for (int v = 0; v < 8; ++v) { sumExp[v] = 0.f; sumS[v] = 0.f; cnt[v] = 0.f; }

#if USE_TDM
  const bool loader = (wave == 0);
  if (loader)
    tdm_load_tile((unsigned)(uintptr_t)&ldsB[0][0], fbf);   // tile j=0
#endif

  for (int j = 0; j < NROWS / 16; ++j) {
    const int buf     = j & 1;
    const int colBase = j * 16;
    const int colLab  = labels[colBase + lr];

#if USE_TDM
    if (loader) __builtin_amdgcn_s_wait_tensorcnt(0);
    __syncthreads();   // publish tile j; retires reads of buffer (j+1)&1
    if (loader && (j + 1 < NROWS / 16))
      tdm_load_tile((unsigned)(uintptr_t)&ldsB[buf ^ 1][0],
                    fbf + (size_t)(colBase + 16) * DIM);
#else
    __syncthreads();
    {  // cooperative fallback copy: 256 threads x 64 B
      const __bf16* src = fbf + (size_t)colBase * DIM;
      for (int t = threadIdx.x; t < TILE_ELEMS / 8; t += 256)
        *(v8bf*)&ldsB[buf][t * 8] = *(const v8bf*)(src + t * 8);
    }
    __syncthreads();
#endif

    // B fragment from LDS: lane = column N=lr, K-contiguous 16 elems
    // at 16*hf within each 32-wide K chunk (same k order as A).
    const __bf16* bt = &ldsB[buf][lr * DIM + 16 * hf];

    // Register double-buffered B + two independent accumulation chains.
    v8f c0 = {}, c1 = {};
    v16bf b0 = *(const v16bf*)(bt + 0 * 32);
    v16bf b1 = *(const v16bf*)(bt + 1 * 32);
#pragma unroll
    for (int kc = 0; kc < 16; kc += 2) {
      v16bf bn0 = b0, bn1 = b1;
      if (kc + 2 < 16) {
        bn0 = *(const v16bf*)(bt + (kc + 2) * 32);
        bn1 = *(const v16bf*)(bt + (kc + 3) * 32);
      }
      c0 = __builtin_amdgcn_wmma_f32_16x16x32_bf16(
               false, afrag[kc],     false, b0, (short)0, c0, false, false);
      c1 = __builtin_amdgcn_wmma_f32_16x16x32_bf16(
               false, afrag[kc + 1], false, b1, (short)0, c1, false, false);
      b0 = bn0;
      b1 = bn1;
    }

    // Epilogue: lane element v is s[M=v+8*hf][N=lr]
#pragma unroll
    for (int v = 0; v < 8; ++v) {
      float s = c0[v] + c1[v];
      sumExp[v] += __expf(s * INV_T);
      if (rowLab[v] == colLab) { sumS[v] += s; cnt[v] += 1.f; }
    }
  }

  // Reduce across the 16 lanes of each half (bits 0..3 of lane id)
#pragma unroll
  for (int v = 0; v < 8; ++v) {
#pragma unroll
    for (int m = 8; m >= 1; m >>= 1) {
      sumExp[v] += __shfl_xor(sumExp[v], m, 32);
      sumS[v]   += __shfl_xor(sumS[v],   m, 32);
      cnt[v]    += __shfl_xor(cnt[v],    m, 32);
    }
  }

  if (lr == 0) {  // lanes 0 and 16: rows {0..7}+8*hf of this block
    float part = 0.f;
#pragma unroll
    for (int v = 0; v < 8; ++v)
      part += __logf(sumExp[v]) - (sumS[v] * INV_T) / cnt[v];
    atomicAdd(acc, part);
  }
}

// ---------------------------------------------------------------------------
__global__ void supcon_zero(float* acc) {
  if (threadIdx.x == 0) acc[0] = 0.f;
}

__global__ void supcon_finalize(const float* __restrict__ acc,
                                float* __restrict__ out) {
  if (threadIdx.x == 0) out[0] = acc[0] * (1.0f / (float)NROWS);
}

// ---------------------------------------------------------------------------
extern "C" void kernel_launch(void* const* d_in, const int* in_sizes, int n_in,
                              void* d_out, int out_size, void* d_ws, size_t ws_size,
                              hipStream_t stream) {
  (void)in_sizes; (void)n_in; (void)out_size; (void)ws_size;

  const float* feat   = (const float*)d_in[0];
  const int*   labels = (const int*)d_in[1];

  __bf16* fbf = (__bf16*)d_ws;                              // 8 MB bf16 features
  float*  acc = (float*)((char*)d_ws + (size_t)NROWS * DIM * sizeof(__bf16));

  supcon_zero<<<1, 32, 0, stream>>>(acc);
  supcon_normalize<<<NROWS / 8, 256, 0, stream>>>(feat, fbf);
  supcon_main<<<(NROWS / 16) / 8, 256, 0, stream>>>(fbf, labels, acc);
  supcon_finalize<<<1, 32, 0, stream>>>(acc, (float*)d_out);
}
